// LSTMAgentModel_89558658056433
// MI455X (gfx1250) — compile-verified, matrix operand order checked by reading, not run on hardware
//
#include <hip/hip_runtime.h>
#include <math.h>

// ---------------------------------------------------------------------------
// Problem constants (from reference): T=1024, S=1024, H=2048, A=128
// ---------------------------------------------------------------------------
constexpr int T_ = 1024;
constexpr int S_ = 1024;
constexpr int H_ = 2048;
constexpr int G_ = 4 * H_;   // 8192 gate rows
constexpr int A_ = 128;
constexpr int NBLK_LSTM = 64;     // persistent blocks (512 waves -> 512 row tiles)
constexpr int SXS = S_ + 8;       // padded LDS row stride (ushorts) to avoid bank conflicts

// ---------------------------------------------------------------------------
// Workspace layout (bytes, 256-aligned). bf16 weight copies (~117 MB) stay
// resident in the 192 MB L2, so each timestep streams weights from L2 not HBM.
// ---------------------------------------------------------------------------
constexpr size_t au(size_t x) { return (x + 255) & ~(size_t)255; }
constexpr size_t OFF_XB   = 0;
constexpr size_t OFF_WIH1 = au(OFF_XB   + (size_t)T_ * S_ * 2);
constexpr size_t OFF_WHH1 = au(OFF_WIH1 + (size_t)G_ * S_ * 2);
constexpr size_t OFF_WIH2 = au(OFF_WHH1 + (size_t)G_ * H_ * 2);
constexpr size_t OFF_WHH2 = au(OFF_WIH2 + (size_t)G_ * H_ * 2);
constexpr size_t OFF_XPRE = au(OFF_WHH2 + (size_t)G_ * H_ * 2);  // [T, 4H] f32
constexpr size_t OFF_B1   = au(OFF_XPRE + (size_t)T_ * G_ * 4);
constexpr size_t OFF_B2   = au(OFF_B1 + (size_t)G_ * 4);
constexpr size_t OFF_H2   = au(OFF_B2 + (size_t)G_ * 4);         // fp32 h2 for head
constexpr size_t OFF_H1B  = au(OFF_H2 + (size_t)H_ * 4);
constexpr size_t OFF_H2B  = au(OFF_H1B + (size_t)H_ * 2);
constexpr size_t OFF_BAR  = au(OFF_H2B + (size_t)H_ * 2);

#define DEV __device__ __forceinline__

typedef __attribute__((ext_vector_type(16))) __bf16 v16bf;
typedef __attribute__((ext_vector_type(8)))  float  v8f;

struct alignas(16) U128 { unsigned int x, y, z, w; };
union Frag16 { v16bf v; unsigned short u[16]; U128 q[2]; };

DEV unsigned short f2bf(float f) {           // fp32 -> bf16 RNE
  unsigned int u = __float_as_uint(f);
  unsigned int r = u + 0x7FFFu + ((u >> 16) & 1u);
  return (unsigned short)(r >> 16);
}
DEV float sigm(float x) { return 1.0f / (1.0f + __expf(-x)); }

// Sense-reversing device-wide barrier (64 resident blocks; s_sleep in spin).
DEV void grid_barrier(int* cnt, int* gen, int nblk) {
  __threadfence();
  __syncthreads();
  if (threadIdx.x == 0) {
    int g0 = __hip_atomic_load(gen, __ATOMIC_RELAXED, __HIP_MEMORY_SCOPE_AGENT);
    int prev = __hip_atomic_fetch_add(cnt, 1, __ATOMIC_ACQ_REL, __HIP_MEMORY_SCOPE_AGENT);
    if (prev == nblk - 1) {
      __hip_atomic_store(cnt, 0, __ATOMIC_RELAXED, __HIP_MEMORY_SCOPE_AGENT);
      __hip_atomic_fetch_add(gen, 1, __ATOMIC_RELEASE, __HIP_MEMORY_SCOPE_AGENT);
    } else {
      while (__hip_atomic_load(gen, __ATOMIC_ACQUIRE, __HIP_MEMORY_SCOPE_AGENT) == g0)
        __builtin_amdgcn_s_sleep(2);
    }
  }
  __syncthreads();
}

// WMMA-based GEMV accumulate: 16 weight rows (per-lane row pointers, chosen by
// the caller) times the h-vector placed in column N=0 of B. ISA fragment
// layout: lane L (r=L&15, g=L>>4) holds two 16B chunks at K=g*8 and K=16+g*8.
// Result: lanes 0/16 hold D[0..7][0] / D[8..15][0] in the 8 accumulator VGPRs.
DEV v8f wmma_gemv_acc(const unsigned short* __restrict__ arow,
                      const unsigned short* __restrict__ hvec,
                      int r, int g, v8f c) {
  for (int k0 = 0; k0 < H_; k0 += 32) {
    Frag16 af, bf_;
    const unsigned short* ap = arow + k0 + g * 8;
    af.q[0] = *(const U128*)ap;
    af.q[1] = *(const U128*)(ap + 16);
    if (r == 0) {
      const unsigned short* bp = hvec + k0 + g * 8;
      bf_.q[0] = *(const U128*)bp;
      bf_.q[1] = *(const U128*)(bp + 16);
    } else {
      for (int i = 0; i < 16; ++i) bf_.u[i] = 0;
    }
    __builtin_prefetch(arow + k0 + 256, 0, 1);   // global_prefetch_b8 ahead
    c = __builtin_amdgcn_wmma_f32_16x16x32_bf16(false, af.v, false, bf_.v,
                                                (short)0, c, false, false);
  }
  return c;
}

// ---------------------------------------------------------------------------
// K1: fp32 -> bf16 conversion, fused biases, zero state, barrier init.
// Re-runs every launch -> graph replay is deterministic.
// ---------------------------------------------------------------------------
__global__ void prep_kernel(const float* __restrict__ x,
                            const float* __restrict__ Wih1,
                            const float* __restrict__ Whh1,
                            const float* __restrict__ bih1,
                            const float* __restrict__ bhh1,
                            const float* __restrict__ Wih2,
                            const float* __restrict__ Whh2,
                            const float* __restrict__ bih2,
                            const float* __restrict__ bhh2,
                            char* __restrict__ ws) {
  unsigned short* xb    = (unsigned short*)(ws + OFF_XB);
  unsigned short* wih1b = (unsigned short*)(ws + OFF_WIH1);
  unsigned short* whh1b = (unsigned short*)(ws + OFF_WHH1);
  unsigned short* wih2b = (unsigned short*)(ws + OFF_WIH2);
  unsigned short* whh2b = (unsigned short*)(ws + OFF_WHH2);
  float* b1 = (float*)(ws + OFF_B1);
  float* b2 = (float*)(ws + OFF_B2);
  float* h2 = (float*)(ws + OFF_H2);
  unsigned short* h1b = (unsigned short*)(ws + OFF_H1B);
  unsigned short* h2b = (unsigned short*)(ws + OFF_H2B);
  int* bar = (int*)(ws + OFF_BAR);

  const size_t stride = (size_t)gridDim.x * blockDim.x;
  const size_t t0 = (size_t)blockIdx.x * blockDim.x + threadIdx.x;

  for (size_t i = t0; i < (size_t)T_ * S_; i += stride) xb[i] = f2bf(x[i]);
  for (size_t i = t0; i < (size_t)G_ * S_; i += stride) wih1b[i] = f2bf(Wih1[i]);
  for (size_t i = t0; i < (size_t)G_ * H_; i += stride) {
    whh1b[i] = f2bf(Whh1[i]);
    wih2b[i] = f2bf(Wih2[i]);
    whh2b[i] = f2bf(Whh2[i]);
  }
  for (size_t i = t0; i < (size_t)G_; i += stride) {
    b1[i] = bih1[i] + bhh1[i];
    b2[i] = bih2[i] + bhh2[i];
  }
  for (size_t i = t0; i < (size_t)H_; i += stride) {
    h2[i] = 0.0f; h1b[i] = 0; h2b[i] = 0;
  }
  if (t0 == 0) { bar[0] = 0; bar[1] = 0; }
}

// ---------------------------------------------------------------------------
// K2: Xpre[T,4H] = x @ W_ih1^T + b1 as a WMMA bf16 GEMM. One 16x16 output
// tile per wave, K=1024. The 16x1024 x-tile is staged in LDS (padded stride).
// ---------------------------------------------------------------------------
__global__ __launch_bounds__(256) void xproj_kernel(char* __restrict__ ws) {
  const unsigned short* xb = (const unsigned short*)(ws + OFF_XB);
  const unsigned short* wb = (const unsigned short*)(ws + OFF_WIH1);
  const float* b1 = (const float*)(ws + OFF_B1);
  float* xpre = (float*)(ws + OFF_XPRE);

  __shared__ unsigned short sx[16 * SXS];   // ~32.3 KB of 320 KB/WGP

  const int w = threadIdx.x >> 5, lane = threadIdx.x & 31;
  const int r = lane & 15, g = lane >> 4;
  const int wid = blockIdx.x * 8 + w;       // 8 waves/block share tileM
  const int tileM = wid >> 9;               // / 512
  const int tileN = wid & 511;

  for (int i = threadIdx.x; i < 16 * (S_ / 8); i += 256) {
    const int row = i >> 7, col = i & 127;       // S_/8 == 128 U128 per row
    ((U128*)(sx + row * SXS))[col] =
        ((const U128*)(xb + ((size_t)tileM * 16 + row) * S_))[col];
  }
  __syncthreads();

  v8f acc = {};
  const unsigned short* brow = wb + (size_t)(tileN * 16 + r) * S_;
  for (int k0 = 0; k0 < S_; k0 += 32) {
    Frag16 af, bf_;
    const unsigned short* ap = sx + r * SXS + k0 + g * 8;  // ds_load_b128
    af.q[0] = *(const U128*)ap;
    af.q[1] = *(const U128*)(ap + 16);
    const unsigned short* bp = brow + k0 + g * 8;
    bf_.q[0] = *(const U128*)bp;
    bf_.q[1] = *(const U128*)(bp + 16);
    acc = __builtin_amdgcn_wmma_f32_16x16x32_bf16(false, af.v, false, bf_.v,
                                                  (short)0, acc, false, false);
  }
  // D layout: lane (r,g) VGPR v -> D[g*8+v][r]
  const int n = tileN * 16 + r;
  const float bias = b1[n];
  for (int v = 0; v < 8; ++v) {
    const int m = tileM * 16 + g * 8 + v;
    xpre[(size_t)m * G_ + n] = acc[v] + bias;
  }
}

// ---------------------------------------------------------------------------
// K3: persistent recurrence, 2 grid barriers per step.
// Wave gw owns hidden units u in [4gw, 4gw+4) of BOTH layers: its 16 A-rows
// are {gate q, unit 4gw+(r&3)} with q = r>>2, i.e. i/f/g/o x 4 units. After
// the WMMA GEMV, lanes 0/16 hold all 16 gate pre-activations -> the cell
// update happens inside the same wave (per-wave LDS gather), and c1/c2 live
// in registers for the whole sequence. Only the h broadcasts need grid sync.
// ---------------------------------------------------------------------------
__global__ __launch_bounds__(256) void lstm_seq_kernel(char* __restrict__ ws) {
  const unsigned short* Whh1b = (const unsigned short*)(ws + OFF_WHH1);
  const unsigned short* Wih2b = (const unsigned short*)(ws + OFF_WIH2);
  const unsigned short* Whh2b = (const unsigned short*)(ws + OFF_WHH2);
  const float* Xpre = (const float*)(ws + OFF_XPRE);
  const float* b2 = (const float*)(ws + OFF_B2);
  float* h2 = (float*)(ws + OFF_H2);
  unsigned short* h1b = (unsigned short*)(ws + OFF_H1B);
  unsigned short* h2b = (unsigned short*)(ws + OFF_H2B);
  int* bar = (int*)(ws + OFF_BAR);

  __shared__ unsigned short sh1[H_];   // 4 KB: h1 broadcast copy
  __shared__ unsigned short sh2[H_];   // 4 KB: h2 broadcast copy
  __shared__ float gsc[8][16];         // per-wave gate scratch

  const int w = threadIdx.x >> 5, lane = threadIdx.x & 31;
  const int r = lane & 15, g = lane >> 4;
  const int gw = blockIdx.x * 8 + w;          // 0..511
  const int ubase = gw * 4;                   // first hidden unit of this wave
  const int q = r >> 2;                       // gate (0=i,1=f,2=g,3=o)
  const int uo = r & 3;                       // unit offset
  const int nblk = gridDim.x;

  const size_t arow_off = ((size_t)q * H_ + ubase + uo) * (size_t)H_;
  const unsigned short* a1 = Whh1b + arow_off;
  const unsigned short* a2 = Wih2b + arow_off;
  const unsigned short* a3 = Whh2b + arow_off;

  // Cell state lives in registers of lanes 0..3; layer-2 bias is loop-invariant.
  float c1reg = 0.0f, c2reg = 0.0f;
  float b2i = 0.0f, b2f = 0.0f, b2g = 0.0f, b2o = 0.0f;
  if (lane < 4) {
    const int u = ubase + lane;
    b2i = b2[0 * H_ + u];
    b2f = b2[1 * H_ + u];
    b2g = b2[2 * H_ + u];
    b2o = b2[3 * H_ + u];
  }

  for (int t = 0; t < T_; ++t) {
    {   // stage h1(t-1), h2(t-1) into LDS
      U128* d1 = (U128*)sh1; U128* d2 = (U128*)sh2;
      const U128* s1 = (const U128*)h1b; const U128* s2 = (const U128*)h2b;
      for (int i = threadIdx.x; i < H_ / 8; i += 256) { d1[i] = s1[i]; d2[i] = s2[i]; }
    }
    __syncthreads();

    // ---- layer 1: gates + cell, fused inside the wave
    v8f acc = {};
    acc = wmma_gemv_acc(a1, sh1, r, g, acc);
    if (r == 0) {                       // lanes 0 and 16 hold column 0
      for (int v = 0; v < 8; ++v) gsc[w][g * 8 + v] = acc[v];
    }
    if (lane < 4) {                     // same wave: LDS ops are in-order
      const int u = ubase + lane;
      const float* xp = Xpre + (size_t)t * G_;
      float i_ = sigm(gsc[w][lane]      + xp[0 * H_ + u]);
      float f_ = sigm(gsc[w][4 + lane]  + xp[1 * H_ + u]);
      float gg = tanhf(gsc[w][8 + lane] + xp[2 * H_ + u]);
      float o_ = sigm(gsc[w][12 + lane] + xp[3 * H_ + u]);
      c1reg = f_ * c1reg + i_ * gg;
      float hn = o_ * tanhf(c1reg);
      h1b[u] = f2bf(hn);
    }
    grid_barrier(bar, bar + 1, nblk);   // publish h1(t)

    {   // restage h1(t)
      U128* d1 = (U128*)sh1; const U128* s1 = (const U128*)h1b;
      for (int i = threadIdx.x; i < H_ / 8; i += 256) d1[i] = s1[i];
    }
    __syncthreads();

    // ---- layer 2: two fused GEMVs + cell, fused inside the wave
    v8f acc2 = {};
    acc2 = wmma_gemv_acc(a2, sh1, r, g, acc2);
    acc2 = wmma_gemv_acc(a3, sh2, r, g, acc2);
    if (r == 0) {
      for (int v = 0; v < 8; ++v) gsc[w][g * 8 + v] = acc2[v];
    }
    if (lane < 4) {
      const int u = ubase + lane;
      float i_ = sigm(gsc[w][lane]      + b2i);
      float f_ = sigm(gsc[w][4 + lane]  + b2f);
      float gg = tanhf(gsc[w][8 + lane] + b2g);
      float o_ = sigm(gsc[w][12 + lane] + b2o);
      c2reg = f_ * c2reg + i_ * gg;
      float hn = o_ * tanhf(c2reg);
      h2b[u] = f2bf(hn);
      h2[u] = hn;                       // fp32 copy for the head
    }
    grid_barrier(bar, bar + 1, nblk);   // publish h2(t)
  }
}

// ---------------------------------------------------------------------------
// K4: out = W_lin @ h2(T-1) + b_lin   (128x2048, trivial; fp32 throughout)
// ---------------------------------------------------------------------------
__global__ void head_kernel(const float* __restrict__ Wlin,
                            const float* __restrict__ blin,
                            const char* __restrict__ ws,
                            float* __restrict__ out) {
  const float* h2 = (const float*)(ws + OFF_H2);
  const int a = blockIdx.x * blockDim.x + threadIdx.x;
  if (a < A_) {
    const float* wr = Wlin + (size_t)a * H_;
    float s = blin[a];
    for (int k = 0; k < H_; ++k) s += wr[k] * h2[k];
    out[a] = s;
  }
}

extern "C" void kernel_launch(void* const* d_in, const int* in_sizes, int n_in,
                              void* d_out, int out_size, void* d_ws, size_t ws_size,
                              hipStream_t stream) {
  (void)in_sizes; (void)n_in; (void)out_size; (void)ws_size;
  const float* x    = (const float*)d_in[0];
  const float* Wih1 = (const float*)d_in[1];
  const float* Whh1 = (const float*)d_in[2];
  const float* bih1 = (const float*)d_in[3];
  const float* bhh1 = (const float*)d_in[4];
  const float* Wih2 = (const float*)d_in[5];
  const float* Whh2 = (const float*)d_in[6];
  const float* bih2 = (const float*)d_in[7];
  const float* bhh2 = (const float*)d_in[8];
  const float* Wlin = (const float*)d_in[9];
  const float* blin = (const float*)d_in[10];
  char* ws = (char*)d_ws;
  float* out = (float*)d_out;

  hipLaunchKernelGGL(prep_kernel, dim3(2048), dim3(256), 0, stream,
                     x, Wih1, Whh1, bih1, bhh1, Wih2, Whh2, bih2, bhh2, ws);
  hipLaunchKernelGGL(xproj_kernel, dim3((T_ / 16) * (G_ / 16) / 8), dim3(256), 0, stream, ws);
  hipLaunchKernelGGL(lstm_seq_kernel, dim3(NBLK_LSTM), dim3(256), 0, stream, ws);
  hipLaunchKernelGGL(head_kernel, dim3(1), dim3(128), 0, stream, Wlin, blin, ws, out);
}
